// SymmetricTransitionDown_30640296689890
// MI455X (gfx1250) — compile-verified, matrix operand order checked by reading, not run on hardware
//
#include <hip/hip_runtime.h>

#define STRIDEV 4
#define BB 4
#define NN 16384
#define EPSV 1e-5f

typedef __attribute__((ext_vector_type(16))) __bf16 v16bf;
typedef __attribute__((ext_vector_type(8)))  __bf16 v8bf;
typedef __attribute__((ext_vector_type(8)))  float  v8f;
typedef __attribute__((ext_vector_type(2)))  float  v2f;

union AF { v16bf v; v8bf h[2]; __bf16 e[16]; };
union CF { v8f  v; float  e[8];  };

// pLDS layout (floats)
#define PSC1 0
#define PSH1 64
#define PSC2 128
#define PSH2 192
#define PW2  256
#define PW1  320   // w1 rows 0,1 (translation), 128 floats
#define PTOT 448

__global__ __launch_bounds__(256) void std_down_kernel(
    const float* __restrict__ points, const float* __restrict__ features,
    const float* __restrict__ w1,
    const float* __restrict__ g1, const float* __restrict__ be1,
    const float* __restrict__ mu1, const float* __restrict__ va1,
    const float* __restrict__ w2, const float* __restrict__ b2,
    const float* __restrict__ w3,
    const float* __restrict__ g2, const float* __restrict__ be2,
    const float* __restrict__ mu2, const float* __restrict__ va2,
    float* __restrict__ out_pts, float* __restrict__ out_feat) {
  // Per-block caches:
  //  - fragLDS: 16 bf16 B-fragments (w1feat kt0/1 x nt0..3, then w3 kt0/1 x
  //    nt0..3) already in v_wmma_f32_16x16x32_bf16 lane layout -> each wave
  //    fetches a fragment with two ds_load_b128, no conversion.
  //  - pLDS: folded BN scale/shift, w2, w1 translation rows.
  __shared__ __align__(32) __bf16 fragLDS[16][32][16];  // 16 KB
  __shared__ float pLDS[PTOT];

  const int tid = threadIdx.x;
  if (tid < 64) {
    float sc1 = g1[tid] * rsqrtf(va1[tid] + EPSV);
    pLDS[PSC1 + tid] = sc1;
    pLDS[PSH1 + tid] = be1[tid] - mu1[tid] * sc1;
    float sc2 = g2[tid] * rsqrtf(va2[tid] + EPSV);
    pLDS[PSC2 + tid] = sc2;
    pLDS[PSH2 + tid] = be2[tid] - mu2[tid] * sc2;
    pLDS[PW2 + tid] = w2[tid];
  } else if (tid < 192) {
    pLDS[PW1 + (tid - 64)] = w1[tid - 64];  // rows 0,1 of w1
  }
  // Build bf16 fragment cache: 512 (frag,lane) slots, 2 per thread.
#pragma unroll
  for (int slot = tid; slot < 512; slot += 256) {
    const int frag = slot >> 5;
    const int ln   = slot & 31;
    const int hif  = ln >> 4;
    const int mat  = frag >> 3;        // 0 = w1 feature part, 1 = w3
    const int kt   = (frag >> 2) & 1;
    const int col  = (frag & 3) * 16 + (ln & 15);
    const float* W = mat ? w3 : (w1 + 2 * 64);
    AF t;
#pragma unroll
    for (int e = 0; e < 16; ++e) {
      int k = 32 * kt + 8 * hif + e + ((e >= 8) ? 8 : 0);
      t.e[e] = (__bf16)W[k * 64 + col];
    }
    *(v8bf*)&fragLDS[frag][ln][0] = t.h[0];
    *(v8bf*)&fragLDS[frag][ln][8] = t.h[1];
  }
  __syncthreads();

  const int wave = tid >> 5;
  const int lane = tid & 31;
  const int m = blockIdx.x * 8 + wave;   // center index 0..16383
  const int b = m >> 12;                 // N/STRIDE = 4096 centers per batch
  const int p = (m & 4095) << 2;         // original point index
  const int s = lane & 15;               // this lane's neighbor row (M dim)
  const int hi = lane >> 4;
  const int l16 = lane & 15;

  // Center point (broadcast) + strided points_out copy.
  const float cx = points[((long)b * NN + p) * 2 + 0];
  const float cy = points[((long)b * NN + p) * 2 + 1];
  if (lane == 0) { out_pts[2 * m] = cx; out_pts[2 * m + 1] = cy; }

  // Circular neighbor for row s: offsets = [-8..-1, 1..8].
  const int off = (s < 8) ? (s - 8) : (s - 7);
  const int nidx = (p + off) & (NN - 1);
  const long grow = (long)b * NN + nidx;
  const float tx = points[grow * 2 + 0] - cx;
  const float ty = points[grow * 2 + 1] - cy;

  // f32 K=4 WMMA A-operand for the translation rank-2 update:
  // lanes 0-15 carry K=0,1 = (tx,ty) of row M=lane; lanes 16-31 carry K=2,3 = 0.
  const v2f a2 = {hi ? 0.f : tx, hi ? 0.f : ty};

  // Load this lane's share of the 16x64 feature A-tile (4 chunks of 8 floats).
  const float* frow = features + grow * 64;
  float ch[4][8];
#pragma unroll
  for (int c = 0; c < 4; ++c) {
    const float4* fp = (const float4*)(frow + 16 * c + 8 * hi);
    float4 a0 = fp[0], a1 = fp[1];
    ch[c][0] = a0.x; ch[c][1] = a0.y; ch[c][2] = a0.z; ch[c][3] = a0.w;
    ch[c][4] = a1.x; ch[c][5] = a1.y; ch[c][6] = a1.z; ch[c][7] = a1.w;
  }
  AF af[2];
#pragma unroll
  for (int kt = 0; kt < 2; ++kt)
#pragma unroll
    for (int i = 0; i < 8; ++i) {
      af[kt].e[i]     = (__bf16)ch[2 * kt][i];
      af[kt].e[8 + i] = (__bf16)ch[2 * kt + 1][i];
    }

  // ---- GEMM1: h = relu(bn1([t,f] @ w1)); accumulate logit partials h @ w2.
  float part[8];
#pragma unroll
  for (int r = 0; r < 8; ++r) part[r] = 0.f;
#pragma unroll
  for (int nt = 0; nt < 4; ++nt) {
    const int col = nt * 16 + l16;
    // Translation rows of w1 as the 4x16 f32 B-operand (K=2,3 zero-padded).
    const float w1r0 = pLDS[PW1 + col], w1r1 = pLDS[PW1 + 64 + col];
    const v2f bt = {hi ? 0.f : w1r0, hi ? 0.f : w1r1};
    CF acc = {};
    acc.v = __builtin_amdgcn_wmma_f32_16x16x4_f32(false, a2, false, bt,
                                                  (short)0, acc.v, false, false);
    AF b0, b1;
    b0.h[0] = *(const v8bf*)&fragLDS[nt][lane][0];
    b0.h[1] = *(const v8bf*)&fragLDS[nt][lane][8];
    b1.h[0] = *(const v8bf*)&fragLDS[4 + nt][lane][0];
    b1.h[1] = *(const v8bf*)&fragLDS[4 + nt][lane][8];
    acc.v = __builtin_amdgcn_wmma_f32_16x16x32_bf16(false, af[0].v, false, b0.v,
                                                    (short)0, acc.v, false, false);
    acc.v = __builtin_amdgcn_wmma_f32_16x16x32_bf16(false, af[1].v, false, b1.v,
                                                    (short)0, acc.v, false, false);
    const float sc = pLDS[PSC1 + col];
    const float sh = pLDS[PSH1 + col];
    const float w2n = pLDS[PW2 + col];
#pragma unroll
    for (int r = 0; r < 8; ++r) {
      float h = fmaxf(acc.e[r] * sc + sh, 0.f);
      part[r] += h * w2n;
    }
  }
  // Reduce logits over the 16 lanes of this half (N dimension).
#pragma unroll
  for (int mk = 1; mk < 16; mk <<= 1)
#pragma unroll
    for (int r = 0; r < 8; ++r) part[r] += __shfl_xor(part[r], mk, 32);

  // Softmax over the 16 neighbor rows (8 per half, combine via xor-16).
  const float b2v = b2[0];
  float mx = -3.4e38f;
#pragma unroll
  for (int r = 0; r < 8; ++r) { part[r] += b2v; mx = fmaxf(mx, part[r]); }
  mx = fmaxf(mx, __shfl_xor(mx, 16, 32));
  float ssum = 0.f, sw[8];
#pragma unroll
  for (int r = 0; r < 8; ++r) { sw[r] = __expf(part[r] - mx); ssum += sw[r]; }
  ssum += __shfl_xor(ssum, 16, 32);
  const float inv = 1.f / ssum;
#pragma unroll
  for (int r = 0; r < 8; ++r) sw[r] *= inv;

  // ---- GEMM2: y = relu(bn2(f @ w3)); out[col] = sum_s sw[s] * y[s,col].
  float outp[4];
#pragma unroll
  for (int nt = 0; nt < 4; ++nt) {
    const int col = nt * 16 + l16;
    AF b0, b1;
    b0.h[0] = *(const v8bf*)&fragLDS[8 + nt][lane][0];
    b0.h[1] = *(const v8bf*)&fragLDS[8 + nt][lane][8];
    b1.h[0] = *(const v8bf*)&fragLDS[12 + nt][lane][0];
    b1.h[1] = *(const v8bf*)&fragLDS[12 + nt][lane][8];
    CF acc = {};
    acc.v = __builtin_amdgcn_wmma_f32_16x16x32_bf16(false, af[0].v, false, b0.v,
                                                    (short)0, acc.v, false, false);
    acc.v = __builtin_amdgcn_wmma_f32_16x16x32_bf16(false, af[1].v, false, b1.v,
                                                    (short)0, acc.v, false, false);
    const float sc = pLDS[PSC2 + col];
    const float sh = pLDS[PSH2 + col];
    float o = 0.f;
#pragma unroll
    for (int r = 0; r < 8; ++r) o += sw[r] * fmaxf(acc.e[r] * sc + sh, 0.f);
    outp[nt] = o;
  }
#pragma unroll
  for (int nt = 0; nt < 4; ++nt) outp[nt] += __shfl_xor(outp[nt], 16, 32);
  if (lane < 16) {
#pragma unroll
    for (int nt = 0; nt < 4; ++nt)
      out_feat[(long)m * 64 + nt * 16 + lane] = outp[nt];
  }
}

extern "C" void kernel_launch(void* const* d_in, const int* in_sizes, int n_in,
                              void* d_out, int out_size, void* d_ws, size_t ws_size,
                              hipStream_t stream) {
  const float* points = (const float*)d_in[0];
  const float* features = (const float*)d_in[1];
  const float* w1  = (const float*)d_in[2];
  const float* g1  = (const float*)d_in[3];
  const float* be1 = (const float*)d_in[4];
  const float* mu1 = (const float*)d_in[5];
  const float* va1 = (const float*)d_in[6];
  const float* w2  = (const float*)d_in[7];
  const float* b2  = (const float*)d_in[8];
  const float* w3  = (const float*)d_in[9];
  const float* g2  = (const float*)d_in[10];
  const float* be2 = (const float*)d_in[11];
  const float* mu2 = (const float*)d_in[12];
  const float* va2 = (const float*)d_in[13];

  float* out = (float*)d_out;
  float* out_pts = out;                                   // (B, N/4, 2) = 32768 floats
  float* out_feat = out + (long)BB * (NN / STRIDEV) * 2;  // (16384, 64)

  const int M = BB * (NN / STRIDEV);  // 16384 centers, 1 wave32 each
  dim3 grid(M / 8), block(256);       // 8 waves per block
  std_down_kernel<<<grid, block, 0, stream>>>(
      points, features, w1, g1, be1, mu1, va1, w2, b2, w3, g2, be2, mu2, va2,
      out_pts, out_feat);
}
